// Head_21964462752435
// MI455X (gfx1250) — matrix operand order repaired and checked by reading, not compiled, not run on hardware
//
#include <hip/hip_runtime.h>
#include <hip/hip_bf16.h>

typedef __attribute__((ext_vector_type(16))) __bf16          bf16x16;
typedef __attribute__((ext_vector_type(16))) unsigned short  u16x16;
typedef __attribute__((ext_vector_type(8)))  unsigned short  u16x8;
typedef __attribute__((ext_vector_type(8)))  float           f32x8;
typedef __attribute__((ext_vector_type(4)))  float           f32x4;

#define WMMA_BF16(A, B, C) \
  __builtin_amdgcn_wmma_f32_16x16x32_bf16(false, (A), false, (B), (short)0, (C), false, false)

// Native f32 -> bf16 conversion: let the compiler use v_cvt_pk_bf16_f32
__device__ __forceinline__ __bf16 f2bf16(float f) { return (__bf16)f; }
__device__ __forceinline__ unsigned short f2bfu(float f) {
  return __builtin_bit_cast(unsigned short, (__bf16)f);
}

// ---------------------------------------------------------------------------
// Kernel 0: W[c][h] fp32 -> Wt[h][c] bf16 (so B-fragments are contiguous in K)
// ---------------------------------------------------------------------------
__global__ void wconv_kernel(const float* __restrict__ Wk,
                             const float* __restrict__ Wq,
                             const float* __restrict__ Wv,
                             unsigned short* __restrict__ WtK,
                             unsigned short* __restrict__ WtQ,
                             unsigned short* __restrict__ WtV) {
  int i = blockIdx.x * 256 + threadIdx.x;                 // 0 .. 1024*64-1
  const float* src = (blockIdx.y == 0) ? Wk : (blockIdx.y == 1) ? Wq : Wv;
  unsigned short* dst = (blockIdx.y == 0) ? WtK : (blockIdx.y == 1) ? WtQ : WtV;
  int c = i >> 6, h = i & 63;
  dst[h * 1024 + c] = f2bfu(src[i]);
}

// ---------------------------------------------------------------------------
// Kernel 1: projections q/k/v = x @ W  via bf16 WMMA, K=1024 in 32 steps.
// One 16-row tile per block; wave w handles N-tile w for all 3 matrices.
// Outputs: Kb,Qb row-major [B*T][64] bf16; Vt transposed [B][64][2048] bf16.
// ---------------------------------------------------------------------------
__global__ void __launch_bounds__(128) proj_kernel(
    const float* __restrict__ x,
    const unsigned short* __restrict__ WtK,
    const unsigned short* __restrict__ WtQ,
    const unsigned short* __restrict__ WtV,
    unsigned short* __restrict__ Kb,
    unsigned short* __restrict__ Qb,
    unsigned short* __restrict__ Vt) {
  const int lane  = threadIdx.x & 31;
  const int wave  = threadIdx.x >> 5;       // N-tile 0..3
  const int hbase = wave * 16;
  const int g0    = blockIdx.x * 16;        // row base over flattened B*T
  const int hi    = lane >> 4;              // K-block selector within lane
  const int n     = lane & 15;

  const float* arow = x + (size_t)(g0 + n) * 1024;
  const int col = hbase + n;
  const unsigned short* bk = WtK + (size_t)col * 1024;
  const unsigned short* bq = WtQ + (size_t)col * 1024;
  const unsigned short* bv = WtV + (size_t)col * 1024;

  f32x8 accK = {}, accQ = {}, accV = {};

  for (int kk = 0; kk < 1024; kk += 32) {
    // A fragment: row (lane&15), 16 contiguous K at kk + 16*hi
    int ao = kk + hi * 16;
    f32x4 f0 = *(const f32x4*)(arow + ao);
    f32x4 f1 = *(const f32x4*)(arow + ao + 4);
    f32x4 f2 = *(const f32x4*)(arow + ao + 8);
    f32x4 f3 = *(const f32x4*)(arow + ao + 12);
    bf16x16 A;
#pragma unroll
    for (int i = 0; i < 4; ++i) {
      A[i]      = f2bf16(f0[i]);
      A[4 + i]  = f2bf16(f1[i]);
      A[8 + i]  = f2bf16(f2[i]);
      A[12 + i] = f2bf16(f3[i]);
    }

    // B fragment: column (lane&15)+hbase, 16 contiguous K at kk + 16*hi
    int kb2 = kk + hi * 16;
    bf16x16 Bk = __builtin_bit_cast(bf16x16, *(const u16x16*)(bk + kb2));
    bf16x16 Bq = __builtin_bit_cast(bf16x16, *(const u16x16*)(bq + kb2));
    bf16x16 Bv = __builtin_bit_cast(bf16x16, *(const u16x16*)(bv + kb2));

    accK = WMMA_BF16(A, Bk, accK);
    accQ = WMMA_BF16(A, Bq, accQ);
    accV = WMMA_BF16(A, Bv, accV);
  }

  // D layout: VGPR r holds row m = r + 8*hi, column n = lane&15
  const int b     = g0 >> 11;               // /2048
  const int tloc0 = (g0 & 2047) + hi * 8;
  u16x8 vp;
#pragma unroll
  for (int r = 0; r < 8; ++r) {
    int m = r + hi * 8;
    size_t o = (size_t)(g0 + m) * 64 + hbase + n;
    Kb[o] = f2bfu(accK[r]);
    Qb[o] = f2bfu(accQ[r]);
    vp[r] = f2bfu(accV[r]);                 // consecutive t -> one packed store
  }
  *(u16x8*)(Vt + (size_t)b * 64 * 2048 + (size_t)(hbase + n) * 2048 + tloc0) = vp;
}

// ---------------------------------------------------------------------------
// Kernel 2: causal flash attention, one 16-query tile per wave.
// Key blocks of 32: S (2 f32 accums, 4 WMMA) -> online softmax (shfl_xor in
// 16-lane groups) -> P through LDS (D-layout -> A-layout) -> O += P.V (4 WMMA)
// ---------------------------------------------------------------------------
__global__ void __launch_bounds__(128) attn_kernel(
    const unsigned short* __restrict__ Qb,
    const unsigned short* __restrict__ Kb,
    const unsigned short* __restrict__ Vt,
    float* __restrict__ out) {
  __shared__ __attribute__((aligned(32))) unsigned short ldsP[4][16][32];

  const int lane  = threadIdx.x & 31;
  const int wave  = threadIdx.x >> 5;
  const int qtile = blockIdx.x * 4 + wave;
  const int g0    = qtile * 16;             // global row base over B*T
  const int b     = g0 >> 11;
  const int q0    = g0 & 2047;              // local t base
  const int hi    = lane >> 4;
  const int n     = lane & 15;

  // Preload Q A-fragments for the two 32-deep K steps over H=64
  bf16x16 QA[2];
  {
    const unsigned short* qrow = Qb + (size_t)(g0 + n) * 64;
#pragma unroll
    for (int ks = 0; ks < 2; ++ks) {
      u16x8 lo = *(const u16x8*)(qrow + ks * 32 + hi * 16);
      u16x8 hh = *(const u16x8*)(qrow + ks * 32 + hi * 16 + 8);
      u16x16 u;
#pragma unroll
      for (int i = 0; i < 8; ++i) { u[i] = lo[i]; u[8 + i] = hh[i]; }
      QA[ks] = __builtin_bit_cast(bf16x16, u);
    }
  }

  float Mrow[8], Lrow[8];
#pragma unroll
  for (int r = 0; r < 8; ++r) { Mrow[r] = -3.0e38f; Lrow[r] = 0.0f; }
  f32x8 accO[4] = {};

  const unsigned short* kbb = Kb + (size_t)b * 2048 * 64;
  const unsigned short* vbb = Vt + (size_t)b * 64 * 2048;

  const int nkb = (q0 + 15) / 32 + 1;       // last block is the only masked one
  for (int kb = 0; kb < nkb; ++kb) {
    const int s0 = kb * 32;

    // ---- S = (Q Kᵀ) * C^-0.5 : two 16x16 f32 accumulators (key cols 0-15 / 16-31)
    f32x8 Sa = {}, Sb = {};
    {
      const unsigned short* kr0 = kbb + (size_t)(s0 + n) * 64 + hi * 16;
      const unsigned short* kr1 = kbb + (size_t)(s0 + 16 + n) * 64 + hi * 16;
      bf16x16 B00 = __builtin_bit_cast(bf16x16, *(const u16x16*)(kr0));
      bf16x16 B01 = __builtin_bit_cast(bf16x16, *(const u16x16*)(kr0 + 32));
      bf16x16 B10 = __builtin_bit_cast(bf16x16, *(const u16x16*)(kr1));
      bf16x16 B11 = __builtin_bit_cast(bf16x16, *(const u16x16*)(kr1 + 32));
      Sa = WMMA_BF16(QA[0], B00, Sa);
      Sa = WMMA_BF16(QA[1], B01, Sa);
      Sb = WMMA_BF16(QA[0], B10, Sb);
      Sb = WMMA_BF16(QA[1], B11, Sb);
    }
    const float sc = 0.03125f;              // 1024^-0.5
#pragma unroll
    for (int r = 0; r < 8; ++r) { Sa[r] *= sc; Sb[r] *= sc; }

    if (kb == nkb - 1) {                    // causal mask, uniform per wave
#pragma unroll
      for (int r = 0; r < 8; ++r) {
        int m = r + hi * 8;
        if (s0 + n      > q0 + m) Sa[r] = -3.0e38f;
        if (s0 + 16 + n > q0 + m) Sb[r] = -3.0e38f;
      }
    }

    // ---- online softmax: rows of D live in fixed VGPR per 16-lane group
    float scl[8];
#pragma unroll
    for (int r = 0; r < 8; ++r) {
      float rm = fmaxf(Sa[r], Sb[r]);
      rm = fmaxf(rm, __shfl_xor(rm, 1, 32));
      rm = fmaxf(rm, __shfl_xor(rm, 2, 32));
      rm = fmaxf(rm, __shfl_xor(rm, 4, 32));
      rm = fmaxf(rm, __shfl_xor(rm, 8, 32));
      float nm = fmaxf(Mrow[r], rm);
      float al = __expf(Mrow[r] - nm);
      float pa = __expf(Sa[r] - nm);
      float pb = __expf(Sb[r] - nm);
      float rs = pa + pb;
      rs += __shfl_xor(rs, 1, 32);
      rs += __shfl_xor(rs, 2, 32);
      rs += __shfl_xor(rs, 4, 32);
      rs += __shfl_xor(rs, 8, 32);
      Lrow[r] = Lrow[r] * al + rs;
      Mrow[r] = nm;
      scl[r]  = al;
      int m = r + hi * 8;
      ldsP[wave][m][n]      = f2bfu(pa);    // D-layout -> [m][s] tile in LDS
      ldsP[wave][m][n + 16] = f2bfu(pb);
    }
#pragma unroll
    for (int j = 0; j < 4; ++j)
#pragma unroll
      for (int r = 0; r < 8; ++r) accO[j][r] *= scl[r];

    // wave-local ordering of LDS store -> load (waves have different trip counts,
    // so no block barrier; DS ops are in-order per wave, wait for data visibility)
    asm volatile("s_wait_dscnt 0" ::: "memory");

    // ---- P as A-fragment (16x32): row = lane&15, K halves per lane[4]
    bf16x16 PA;
    {
      u16x8 lo = *(const u16x8*)&ldsP[wave][n][hi * 16];
      u16x8 hh = *(const u16x8*)&ldsP[wave][n][hi * 16 + 8];
      u16x16 u;
#pragma unroll
      for (int i = 0; i < 8; ++i) { u[i] = lo[i]; u[8 + i] = hh[i]; }
      PA = __builtin_bit_cast(bf16x16, u);
    }

    // ---- O += P.V  (V transposed: column h contiguous in s)
#pragma unroll
    for (int j = 0; j < 4; ++j) {
      const unsigned short* vr = vbb + (size_t)(j * 16 + n) * 2048 + s0 + hi * 16;
      bf16x16 BV = __builtin_bit_cast(bf16x16, *(const u16x16*)vr);
      accO[j] = WMMA_BF16(PA, BV, accO[j]);
    }
  }

  // ---- normalize and write fp32 output
#pragma unroll
  for (int r = 0; r < 8; ++r) {
    float inv = 1.0f / Lrow[r];
    int m = r + hi * 8;
    float* orow = out + (size_t)(g0 + m) * 64;
#pragma unroll
    for (int j = 0; j < 4; ++j) orow[j * 16 + n] = accO[j][r] * inv;
  }
}

// ---------------------------------------------------------------------------
extern "C" void kernel_launch(void* const* d_in, const int* in_sizes, int n_in,
                              void* d_out, int out_size, void* d_ws, size_t ws_size,
                              hipStream_t stream) {
  const float* x  = (const float*)d_in[0];
  const float* Wk = (const float*)d_in[1];
  const float* Wq = (const float*)d_in[2];
  const float* Wv = (const float*)d_in[3];
  float* out = (float*)d_out;

  // workspace layout (bf16), all offsets 256B-aligned:
  //   Wt{K,Q,V}: 64*1024 each (128 KiB each)
  //   Kb, Qb   : [B*T][64]      (2 MiB each)
  //   Vt       : [B][64][2048]  (2 MiB)
  char* ws = (char*)d_ws;
  unsigned short* WtK = (unsigned short*)(ws);
  unsigned short* WtQ = (unsigned short*)(ws + (size_t)131072);
  unsigned short* WtV = (unsigned short*)(ws + (size_t)262144);
  unsigned short* Kb  = (unsigned short*)(ws + (size_t)393216);
  unsigned short* Qb  = (unsigned short*)(ws + (size_t)393216 + (size_t)2097152);
  unsigned short* Vt  = (unsigned short*)(ws + (size_t)393216 + (size_t)2 * 2097152);

  wconv_kernel<<<dim3(256, 3), 256, 0, stream>>>(Wk, Wq, Wv, WtK, WtQ, WtV);
  proj_kernel<<<dim3(1024), 128, 0, stream>>>(x, WtK, WtQ, WtV, Kb, Qb, Vt);
  attn_kernel<<<dim3(256), 128, 0, stream>>>(Qb, Kb, Vt, out);
}
